// CFGOracle_81947976007921
// MI455X (gfx1250) — matrix operand order repaired and checked by reading, not compiled, not run on hardware
//
#include <hip/hip_runtime.h>
#include <hip/hip_bf16.h>

typedef __attribute__((ext_vector_type(16))) _Float16 v16h;
typedef __attribute__((ext_vector_type(8)))  _Float16 v8h;
typedef __attribute__((ext_vector_type(8)))  float    v8f;

#define BB   16
#define NN   24
#define TT   32
#define KK   512
#define HH   128
#define VV   32000
#define NEGV (-1e9f)
#define EPSV (1e-38f)

#define SHUF16(lo, hi) __builtin_shufflevector(lo, hi, 0,1,2,3,4,5,6,7,8,9,10,11,12,13,14,15)

// ---------------- LayerNorm + MLP + log_softmax -> Phi[b][n][t] ----------------
__global__ void k_mlp_phi(const float* __restrict__ g_seq,
                          const float* __restrict__ ln_g, const float* __restrict__ ln_b,
                          const float* __restrict__ W1, const float* __restrict__ b1,
                          const float* __restrict__ W2, const float* __restrict__ b2,
                          float* __restrict__ Phi) {
  int bt = blockIdx.x; int b = bt / TT; int t = bt % TT;
  int tid = threadIdx.x; // 128 threads
  __shared__ float g[KK];
  __shared__ float red[HH];
  __shared__ float h[HH];
  __shared__ float o[NN];
  const float* row = g_seq + (size_t)bt * KK;
  float s = 0.f;
  for (int k = tid; k < KK; k += HH) { float v = row[k]; g[k] = v; s += v; }
  red[tid] = s; __syncthreads();
  for (int off = 64; off > 0; off >>= 1) { if (tid < off) red[tid] += red[tid+off]; __syncthreads(); }
  float mu = red[0] / (float)KK; __syncthreads();
  float s2 = 0.f;
  for (int k = tid; k < KK; k += HH) { float d = g[k]-mu; s2 += d*d; }
  red[tid] = s2; __syncthreads();
  for (int off = 64; off > 0; off >>= 1) { if (tid < off) red[tid] += red[tid+off]; __syncthreads(); }
  float inv = rsqrtf(red[0]/(float)KK + 1e-5f); __syncthreads();
  for (int k = tid; k < KK; k += HH) g[k] = (g[k]-mu)*inv*ln_g[k] + ln_b[k];
  __syncthreads();
  float acc = b1[tid];
  for (int k = 0; k < KK; ++k) acc += g[k]*W1[k*HH + tid];
  h[tid] = 0.5f*acc*(1.f + erff(acc*0.7071067811865475f)); // exact gelu
  __syncthreads();
  if (tid < NN) {
    float a = b2[tid];
    for (int j = 0; j < HH; ++j) a += h[j]*W2[j*NN + tid];
    o[tid] = a;
  }
  __syncthreads();
  if (tid == 0) {
    float m = o[0]; for (int n = 1; n < NN; ++n) m = fmaxf(m, o[n]);
    float se = 0.f; for (int n = 0; n < NN; ++n) se += expf(o[n]-m);
    red[0] = m + logf(se);
  }
  __syncthreads();
  if (tid < NN) Phi[((size_t)b*NN + tid)*TT + t] = o[tid] - red[0];
}

// ---------------- theta log_softmax over z ----------------
__global__ void k_theta(const float* __restrict__ Theta, float* __restrict__ tl) {
  int r = blockIdx.x; int tid = threadIdx.x;
  __shared__ float v[NN]; __shared__ float ml[2];
  if (tid < NN) v[tid] = Theta[r*NN + tid];
  __syncthreads();
  if (tid == 0) {
    float m = v[0]; for (int z = 1; z < NN; ++z) m = fmaxf(m, v[z]);
    float se = 0.f; for (int z = 0; z < NN; ++z) se += expf(v[z]-m);
    ml[0] = m; ml[1] = logf(se);
  }
  __syncthreads();
  if (tid < NN) tl[r*NN + tid] = v[tid] - ml[0] - ml[1];
}

__global__ void k_tm(const float* __restrict__ tl, float* __restrict__ tm) {
  __shared__ float red[256];
  float m = -3.4e38f;
  for (int i = threadIdx.x; i < NN*NN*NN; i += 256) m = fmaxf(m, tl[i]);
  red[threadIdx.x] = m; __syncthreads();
  for (int off = 128; off > 0; off >>= 1) {
    if (threadIdx.x < off) red[threadIdx.x] = fmaxf(red[threadIdx.x], red[threadIdx.x+off]);
    __syncthreads();
  }
  if (threadIdx.x == 0) tm[0] = red[0];
}

// et = exp(theta_log - tm) in three permutations (f16), 32 rows x K=576 (rows >=24 pre-zeroed)
__global__ void k_et(const float* __restrict__ tl, const float* __restrict__ tm,
                     _Float16* __restrict__ etI, _Float16* __restrict__ etL,
                     _Float16* __restrict__ etR) {
  int r = blockIdx.x; int x = r / NN, y = r % NN; int z = threadIdx.x;
  if (z < NN) {
    _Float16 h = (_Float16)expf(tl[(x*NN+y)*NN + z] - tm[0]);
    etI[x*576 + y*NN + z] = h; // inner:  [x][yz]
    etL[y*576 + x*NN + z] = h; // cL:     [y][xz]
    etR[z*576 + x*NN + y] = h; // cR:     [z][xy]
  }
}

// ---------------- beta / alpha init ----------------
__global__ void k_beta_init(const float* __restrict__ Phi, float* __restrict__ beta) {
  int i = blockIdx.x*blockDim.x + threadIdx.x;
  if (i >= BB*NN*TT*TT) return;
  int t2 = i & 31; int t1 = (i >> 5) & 31; int n = (i >> 10) % NN; int b = i / (NN*TT*TT);
  beta[i] = (t1 == t2) ? Phi[((size_t)b*NN + n)*TT + t1] : NEGV;
}

__global__ void k_alpha_init(const float* __restrict__ beta, float* __restrict__ alpha,
                             float* __restrict__ logZ) {
  int i = blockIdx.x*blockDim.x + threadIdx.x;
  if (i >= BB*NN*TT*TT) return;
  int t2 = i & 31; int t1 = (i >> 5) & 31; int n = (i >> 10) % NN; int b = i / (NN*TT*TT);
  bool root = (n == 0) && (t1 == 0) && (t2 == TT-1);
  alpha[i] = root ? 0.f : NEGV;
  if (root) logZ[b] = beta[i];
}

// -------- inside: build Kron el(x)er in [b][col][k] layout (k contiguous), pad cols --------
__global__ void k_inside_prep(const float* __restrict__ beta, _Float16* __restrict__ kron,
                              float* __restrict__ ml_ws, float* __restrict__ mr_ws,
                              int l, int P, int S, int PSpad) {
  int blk = blockIdx.x; int b = blk / PSpad; int ps = blk % PSpad;
  int tid = threadIdx.x;
  _Float16* krow = kron + ((size_t)b*256 + ps)*576;
  if (ps >= P*S) {  // zero pad column so GEMM loads need no guard
    for (int idx = tid; idx < 576; idx += 64) krow[idx] = (_Float16)0.f;
    return;
  }
  int i = ps / S; int s = ps % S;
  __shared__ float L[NN], R[NN], mm[2];
  for (int t = tid; t < 2*NN; t += 64) {
    if (t < NN) L[t]    = beta[(((size_t)b*NN + t)*TT + i)*TT + (i+s)];
    else        R[t-NN] = beta[(((size_t)b*NN + (t-NN))*TT + (i+s+1))*TT + (i+l-1)];
  }
  __syncthreads();
  if (tid == 0) { float m=L[0]; for(int y=1;y<NN;++y)m=fmaxf(m,L[y]); mm[0]=m; ml_ws[b*(P*S)+ps]=m; }
  if (tid == 1) { float m=R[0]; for(int z=1;z<NN;++z)m=fmaxf(m,R[z]); mm[1]=m; mr_ws[b*(P*S)+ps]=m; }
  __syncthreads();
  float ml = mm[0], mr = mm[1];
  for (int idx = tid; idx < 576; idx += 64) {
    int y = idx / NN, z = idx % NN;
    krow[idx] = (_Float16)(expf(L[y]-ml)*expf(R[z]-mr));
  }
}

// -------- WMMA GEMM: inner[b][m][col] = etm(32x576, zero-padded) @ kron[b][col][k] --------
__global__ void k_gemm24(const _Float16* __restrict__ etm, const _Float16* __restrict__ kron,
                         float* __restrict__ inner, int PS) {
  int ct = blockIdx.x;        // column tile
  int xt = blockIdx.y;        // row tile (0: M 0-15, 1: M 16-31, rows>=24 are zeros)
  int b  = blockIdx.z;
  int lane = threadIdx.x;     // one wave32 per block
  int m   = xt*16 + (lane & 15);
  int col = ct*16 + (lane & 15);
  const _Float16* arow = etm + (size_t)m*576;
  const _Float16* brow = kron + ((size_t)b*256 + col)*576;
  int koffA = (lane < 16) ? 0 : 8;   // 16-bit A layout: K 0-7/16-23 vs 8-15/24-31
  int koffB = (lane < 16) ? 0 : 8;   // B layout: lane=column, K 0-7/16-23 vs 8-15/24-31 (matches A)
  __builtin_prefetch(brow, 0, 1);            // global_prefetch_b8
  __builtin_prefetch(brow + 512, 0, 1);
  v8f c = {};
#pragma unroll
  for (int kk = 0; kk < 18; ++kk) {          // K = 576 = 18 * 32
    int kbase = kk*32;
    v8h alo = *(const v8h*)(arow + kbase + koffA);
    v8h ahi = *(const v8h*)(arow + kbase + 16 + koffA);
    v8h blo = *(const v8h*)(brow + kbase + koffB);
    v8h bhi = *(const v8h*)(brow + kbase + 16 + koffB);
    v16h av = SHUF16(alo, ahi);
    v16h bv = SHUF16(blo, bhi);
    c = __builtin_amdgcn_wmma_f32_16x16x32_f16(false, av, false, bv, (short)0, c, false, false);
  }
  int rowbase = xt*16 + 8*(lane >> 4);
#pragma unroll
  for (int j = 0; j < 8; ++j) {
    int mm = rowbase + j;
    if (mm < NN && col < PS) inner[((size_t)b*NN + mm)*256 + col] = c[j];
  }
}

// ---------------- inside: logsumexp over splits -> beta[b][x][i][i+l-1] ----------------
__global__ void k_inside_reduce(const float* __restrict__ inner, const float* __restrict__ ml_ws,
                                const float* __restrict__ mr_ws, const float* __restrict__ tm,
                                float* __restrict__ beta, int l, int P, int S) {
  int idx = blockIdx.x*blockDim.x + threadIdx.x;
  if (idx >= BB*NN*P) return;
  int i = idx % P; int x = (idx / P) % NN; int b = idx / (NN*P);
  float tmv = tm[0];
  float sc[TT-1];
  float mmax = -3.4e38f;
  for (int s = 0; s < S; ++s) {
    float v = inner[((size_t)b*NN + x)*256 + i*S + s];
    float t = logf(fmaxf(v, EPSV)) + tmv + ml_ws[b*(P*S)+i*S+s] + mr_ws[b*(P*S)+i*S+s];
    sc[s] = t; mmax = fmaxf(mmax, t);
  }
  float se = 0.f;
  for (int s = 0; s < S; ++s) se += expf(sc[s]-mmax);
  beta[(((size_t)b*NN + x)*TT + i)*TT + (i+l-1)] = mmax + logf(se);
}

// -------- outside: build Kron ea(x)erb and ea(x)elb, [b][col][k] layout, pad cols --------
__global__ void k_outside_prep(const float* __restrict__ beta, const float* __restrict__ alpha,
                               _Float16* __restrict__ kronL, _Float16* __restrict__ kronR,
                               float* __restrict__ ma_ws, float* __restrict__ mlb_ws,
                               float* __restrict__ mrb_ws, int l, int P, int S, int PSpad) {
  int blk = blockIdx.x; int b = blk / PSpad; int ps = blk % PSpad;
  int tid = threadIdx.x;
  _Float16* krowL = kronL + ((size_t)b*256 + ps)*576;
  _Float16* krowR = kronR + ((size_t)b*256 + ps)*576;
  if (ps >= P*S) {
    for (int idx = tid; idx < 576; idx += 64) { krowL[idx] = (_Float16)0.f; krowR[idx] = (_Float16)0.f; }
    return;
  }
  int i = ps / S; int s = ps % S; int j = i + l - 1;
  __shared__ float A[NN], Lb[NN], Rb[NN], mx[3];
  for (int t = tid; t < 3*NN; t += 64) {
    if (t < NN)        A[t]       = alpha[(((size_t)b*NN + t)*TT + i)*TT + j];
    else if (t < 2*NN) Lb[t-NN]   = beta[(((size_t)b*NN + (t-NN))*TT + i)*TT + (i+s)];
    else               Rb[t-2*NN] = beta[(((size_t)b*NN + (t-2*NN))*TT + (i+s+1))*TT + j];
  }
  __syncthreads();
  if (tid == 0) { float m=A[0];  for(int x=1;x<NN;++x)m=fmaxf(m,A[x]);  mx[0]=m; ma_ws[b*(P*S)+ps]=m; }
  if (tid == 1) { float m=Lb[0]; for(int y=1;y<NN;++y)m=fmaxf(m,Lb[y]); mx[1]=m; mlb_ws[b*(P*S)+ps]=m; }
  if (tid == 2) { float m=Rb[0]; for(int z=1;z<NN;++z)m=fmaxf(m,Rb[z]); mx[2]=m; mrb_ws[b*(P*S)+ps]=m; }
  __syncthreads();
  float ma = mx[0], mlb = mx[1], mrb = mx[2];
  for (int idx = tid; idx < 576; idx += 64) {
    int x = idx / NN, w = idx % NN;
    float ea = expf(A[x]-ma);
    krowL[idx] = (_Float16)(ea*expf(Rb[w]-mrb)); // [xz]
    krowR[idx] = (_Float16)(ea*expf(Lb[w]-mlb)); // [xy]
  }
}

// ---------------- outside: logaddexp scatter into alpha ----------------
__global__ void k_outside_update(const float* __restrict__ inner, const float* __restrict__ ma_ws,
                                 const float* __restrict__ mh_ws, const float* __restrict__ tm,
                                 float* __restrict__ alpha, int l, int P, int S, int isR) {
  int idx = blockIdx.x*blockDim.x + threadIdx.x;
  if (idx >= BB*NN*P*S) return;
  int s = idx % S; int i = (idx / S) % P; int y = (idx / (S*P)) % NN; int b = idx / (NN*P*S);
  float v  = inner[((size_t)b*NN + y)*256 + i*S + s];
  float sc = logf(fmaxf(v, EPSV)) + tm[0] + ma_ws[b*(P*S)+i*S+s] + mh_ws[b*(P*S)+i*S+s];
  size_t cell = isR ? ((((size_t)b*NN + y)*TT + (i+s+1))*TT + (i+l-1))
                    : ((((size_t)b*NN + y)*TT + i)*TT + (i+s));
  float old = alpha[cell];
  float mmax = fmaxf(old, sc);
  alpha[cell] = mmax + log1pf(expf(fminf(old, sc) - mmax));
}

// ------- p_nt staged as f16 [bt][32] (cols>=24 zero) for the WMMA mask GEMM -------
__global__ void k_pnt(const float* __restrict__ alpha, const float* __restrict__ beta,
                      const float* __restrict__ logZ, _Float16* __restrict__ pnth) {
  int idx = blockIdx.x*blockDim.x + threadIdx.x;
  if (idx >= BB*TT*32) return;
  int n = idx & 31; int t = (idx >> 5) % TT; int b = idx / (TT*32);
  float v = 0.f;
  if (n < NN) {
    size_t cell = (((size_t)b*NN + n)*TT + t)*TT + t;
    v = expf(alpha[cell] + beta[cell] - logZ[b]);
  }
  pnth[idx] = (_Float16)v;
}

// ------- nt2vocab transposed + padded to f16 [v][32] (rows>=24 zero) -------
__global__ void k_vocpad(const float* __restrict__ voc, _Float16* __restrict__ vocc) {
  int idx = blockIdx.x*blockDim.x + threadIdx.x;
  if (idx >= VV*32) return;
  int v = idx % VV; int k = idx / VV;  // consecutive threads -> consecutive v (coalesced reads)
  vocc[(size_t)v*32 + k] = (k < NN) ? (_Float16)voc[(size_t)k*VV + v] : (_Float16)0.f;
}

// ------- mask_logits = log(1e-6 + pnt^T @ nt2vocab) via WMMA (K=32, one step) -------
__global__ void k_mask(const _Float16* __restrict__ pnth, const _Float16* __restrict__ vocc,
                       float* __restrict__ out) {
  int lane = threadIdx.x;  // one wave32 per block
  int vt = blockIdx.x;     // vocab tile (2000)
  int rt = blockIdx.y;     // (b,t) row tile (32)
  int r   = rt*16 + (lane & 15);
  int col = vt*16 + (lane & 15);
  int koffA = (lane < 16) ? 0 : 8;
  int koffB = (lane < 16) ? 0 : 8;
  const _Float16* arow = pnth + (size_t)r*32;
  const _Float16* brow = vocc + (size_t)col*32;
  v8h alo = *(const v8h*)(arow + koffA);
  v8h ahi = *(const v8h*)(arow + 16 + koffA);
  v8h blo = *(const v8h*)(brow + koffB);
  v8h bhi = *(const v8h*)(brow + 16 + koffB);
  v16h av = SHUF16(alo, ahi);
  v16h bv = SHUF16(blo, bhi);
  v8f c = {};
  c = __builtin_amdgcn_wmma_f32_16x16x32_f16(false, av, false, bv, (short)0, c, false, false);
  int rowbase = rt*16 + 8*(lane >> 4);
#pragma unroll
  for (int j = 0; j < 8; ++j) {
    int rr = rowbase + j;
    out[(size_t)rr*VV + col] = logf(1e-6f + c[j]);
  }
}

__global__ void k_loss(const float* __restrict__ logZ, float* __restrict__ out) {
  if (threadIdx.x == 0) {
    float s = 0.f;
    for (int b = 0; b < BB; ++b) s += logZ[b];
    out[0] = -s / (float)BB;
  }
}

extern "C" void kernel_launch(void* const* d_in, const int* in_sizes, int n_in,
                              void* d_out, int out_size, void* d_ws, size_t ws_size,
                              hipStream_t stream) {
  (void)in_sizes; (void)n_in; (void)out_size; (void)ws_size;
  const float* g_seq    = (const float*)d_in[0];
  const float* Theta    = (const float*)d_in[1];
  const float* nt2vocab = (const float*)d_in[2];
  const float* ln_gamma = (const float*)d_in[3];
  const float* ln_beta  = (const float*)d_in[4];
  const float* W1       = (const float*)d_in[5];
  const float* b1       = (const float*)d_in[6];
  const float* W2       = (const float*)d_in[7];
  const float* b2       = (const float*)d_in[8];
  float* out = (float*)d_out;

  char* w = (char*)d_ws; size_t off = 0;
  auto carve = [&](size_t bytes) -> void* {
    void* p = w + off; off = (off + bytes + 255) & ~(size_t)255; return p;
  };
  float*    Phi    = (float*)   carve((size_t)BB*NN*TT*4);
  float*    tl     = (float*)   carve((size_t)NN*NN*NN*4);
  float*    tm     = (float*)   carve(4);
  _Float16* etI    = (_Float16*)carve((size_t)32*576*2);      // rows 24..31 zero
  _Float16* etL    = (_Float16*)carve((size_t)32*576*2);
  _Float16* etR    = (_Float16*)carve((size_t)32*576*2);
  float*    beta   = (float*)   carve((size_t)BB*NN*TT*TT*4);
  float*    alpha  = (float*)   carve((size_t)BB*NN*TT*TT*4);
  float*    logZ   = (float*)   carve((size_t)BB*4);
  float*    ml     = (float*)   carve((size_t)BB*256*4);
  float*    mr     = (float*)   carve((size_t)BB*256*4);
  float*    ma     = (float*)   carve((size_t)BB*256*4);
  float*    mlb    = (float*)   carve((size_t)BB*256*4);
  float*    mrb    = (float*)   carve((size_t)BB*256*4);
  _Float16* kronA  = (_Float16*)carve((size_t)BB*256*576*2);  // [b][col][k]
  _Float16* kronB  = (_Float16*)carve((size_t)BB*256*576*2);
  float*    innerA = (float*)   carve((size_t)BB*NN*256*4);
  float*    innerB = (float*)   carve((size_t)BB*NN*256*4);
  _Float16* pnth   = (_Float16*)carve((size_t)BB*TT*32*2);    // [bt][32], cols>=24 zero
  _Float16* vocc   = (_Float16*)carve((size_t)VV*32*2);       // [v][32], rows>=24 zero

  // Front-end
  k_mlp_phi<<<BB*TT, HH, 0, stream>>>(g_seq, ln_gamma, ln_beta, W1, b1, W2, b2, Phi);
  k_theta<<<NN*NN, 32, 0, stream>>>(Theta, tl);
  k_tm<<<1, 256, 0, stream>>>(tl, tm);
  hipMemsetAsync(etI, 0, (size_t)32*576*2, stream);
  hipMemsetAsync(etL, 0, (size_t)32*576*2, stream);
  hipMemsetAsync(etR, 0, (size_t)32*576*2, stream);
  k_et<<<NN*NN, 32, 0, stream>>>(tl, tm, etI, etL, etR);
  k_beta_init<<<(BB*NN*TT*TT + 255)/256, 256, 0, stream>>>(Phi, beta);
  k_vocpad<<<(VV*32 + 255)/256, 256, 0, stream>>>(nt2vocab, vocc);

  // Inside pass
  for (int l = 2; l <= TT; ++l) {
    int P = TT - l + 1, S = l - 1, PS = P*S;
    int ctiles = (PS + 15) / 16, PSpad = ctiles*16;
    k_inside_prep<<<BB*PSpad, 64, 0, stream>>>(beta, kronA, ml, mr, l, P, S, PSpad);
    k_gemm24<<<dim3(ctiles, 2, BB), 32, 0, stream>>>(etI, kronA, innerA, PS);
    k_inside_reduce<<<(BB*NN*P + 255)/256, 256, 0, stream>>>(innerA, ml, mr, tm, beta, l, P, S);
  }

  k_alpha_init<<<(BB*NN*TT*TT + 255)/256, 256, 0, stream>>>(beta, alpha, logZ);

  // Outside pass
  for (int l = TT; l >= 2; --l) {
    int P = TT - l + 1, S = l - 1, PS = P*S;
    int ctiles = (PS + 15) / 16, PSpad = ctiles*16;
    k_outside_prep<<<BB*PSpad, 64, 0, stream>>>(beta, alpha, kronA, kronB, ma, mlb, mrb, l, P, S, PSpad);
    k_gemm24<<<dim3(ctiles, 2, BB), 32, 0, stream>>>(etL, kronA, innerA, PS);
    k_gemm24<<<dim3(ctiles, 2, BB), 32, 0, stream>>>(etR, kronB, innerB, PS);
    k_outside_update<<<(BB*NN*PS + 255)/256, 256, 0, stream>>>(innerA, ma, mrb, tm, alpha, l, P, S, 0);
    k_outside_update<<<(BB*NN*PS + 255)/256, 256, 0, stream>>>(innerB, ma, mlb, tm, alpha, l, P, S, 1);
  }

  // Back-end
  k_pnt<<<(BB*TT*32 + 255)/256, 256, 0, stream>>>(alpha, beta, logZ, pnth);
  k_mask<<<dim3(VV/16, (BB*TT)/16), 32, 0, stream>>>(pnth, vocc, out);
  k_loss<<<1, 32, 0, stream>>>(logZ, out + (size_t)BB*TT*VV);
}
